// EnhancedTemporalAttention_2748779069611
// MI455X (gfx1250) — matrix operand order repaired and checked by reading, not compiled, not run on hardware
//
#include <hip/hip_runtime.h>

// ---------------------------------------------------------------------------
// Problem constants (reference: B=16, T=2048, F=1024)
// ---------------------------------------------------------------------------
#define B_DIM 16
#define T_DIM 2048
#define F_DIM 1024
#define BT_DIM (B_DIM * T_DIM)            // 32768
#define BTF ((size_t)BT_DIM * F_DIM)      // 33554432
#define LDA 40                            // K-tile LDS row stride (80B, conflict-free)
#define QLD 1032                          // resident-Q LDS row stride (2064B = 4 banks/row)
#define Q_ELEMS (128 * QLD)               // 132096 bf16
#define KS_ELEMS (128 * LDA)              // 5120 bf16 per K buffer
#define ATTN_SMEM_BYTES ((Q_ELEMS + 2 * KS_ELEMS) * 2)   // 284672 B (< 320KB WGP LDS)
#define SCALE_QK 0.03125f                 // 1/sqrt(1024)

typedef __bf16 bf16_t;
typedef bf16_t v16bf __attribute__((ext_vector_type(16)));
typedef bf16_t v8bf  __attribute__((ext_vector_type(8)));
typedef bf16_t v2bf  __attribute__((ext_vector_type(2)));
typedef float  v8f   __attribute__((ext_vector_type(8)));
typedef int    v4i   __attribute__((vector_size(4 * sizeof(int))));

#if defined(__AMDGCN__) && __has_builtin(__builtin_amdgcn_global_load_async_to_lds_b128)
#define USE_ASYNC_LDS 1
#else
#define USE_ASYNC_LDS 0
#endif

// ---------------------------------------------------------------------------
// Helpers
// ---------------------------------------------------------------------------
__device__ __forceinline__ bf16_t f2bf(float f) {
  union { float f; unsigned u; } in; in.f = f;
  unsigned r = in.u + 0x7FFFu + ((in.u >> 16) & 1u);   // round-to-nearest-even
  union { unsigned short s; bf16_t b; } out;
  out.s = (unsigned short)(r >> 16);
  return out.b;
}

// two floats -> packed 2xbf16 (v_cvt_pk_bf16_f32 when available)
__device__ __forceinline__ unsigned f2bf2(float a, float b) {
#if defined(__AMDGCN__) && __has_builtin(__builtin_amdgcn_cvt_pk_bf16_f32)
  union { v2bf v; unsigned u; } c;
  c.v = __builtin_amdgcn_cvt_pk_bf16_f32(a, b);
  return c.u;
#else
  union { unsigned short s[2]; unsigned u; } c;
  union { bf16_t b; unsigned short s; } x0, x1;
  x0.b = f2bf(a); x1.b = f2bf(b);
  c.s[0] = x0.s; c.s[1] = x1.s;
  return c.u;
#endif
}

__device__ __forceinline__ float bf2f(bf16_t b) {
  union { bf16_t b; unsigned short s; } in; in.b = b;
  union { unsigned u; float f; } out;
  out.u = ((unsigned)in.s) << 16;
  return out.f;
}

__device__ __forceinline__ v8f zero8() {
  v8f z = {0.f, 0.f, 0.f, 0.f, 0.f, 0.f, 0.f, 0.f};
  return z;
}

__device__ __forceinline__ v8f wmma_bf16(v16bf a, v16bf b, v8f c) {
  return __builtin_amdgcn_wmma_f32_16x16x32_bf16(
      /*neg_a=*/false, a, /*neg_b=*/false, b,
      /*c_mod=*/(short)0, c, /*reuse_a=*/false, /*reuse_b=*/false);
}

// 16B global -> LDS copy; async (ASYNCcnt) when available, else through VGPRs.
__device__ __forceinline__ void cp_b128(bf16_t* dst, const bf16_t* src) {
#if USE_ASYNC_LDS
  __builtin_amdgcn_global_load_async_to_lds_b128(
      (v4i*)src, (v4i*)dst, /*offset=*/0, /*cpol=*/0);
#else
  *(v8bf*)dst = *(const v8bf*)src;
#endif
}

__device__ __forceinline__ void wait_async() {
#if USE_ASYNC_LDS
#if __has_builtin(__builtin_amdgcn_s_wait_asynccnt)
  __builtin_amdgcn_s_wait_asynccnt(0);
#else
  asm volatile("s_wait_asynccnt 0x0" ::: "memory");
#endif
#endif
}

// A fragment (16x32 bf16): lane halves hold K chunks {base..+7, base+16..+23}
__device__ __forceinline__ v16bf load_a_frag_s(const bf16_t* As, int stride,
                                               int m0, int kofs, int lane) {
  const int row = m0 + (lane & 15);
  const int kb  = kofs + (lane >> 4) * 8;
  const bf16_t* p = As + row * stride + kb;
  union { v16bf v; v8bf h[2]; } f;
  f.h[0] = *(const v8bf*)(p);
  f.h[1] = *(const v8bf*)(p + 16);
  return f.v;
}

// B fragment (32x16 bf16) from [N][K] LDS tile: lane = col, K chunks {base..+7, base+16..+23}
__device__ __forceinline__ v16bf load_b_frag(const bf16_t* Bs, int n0, int lane) {
  const int nr = n0 + (lane & 15);
  const int kb = (lane >> 4) * 8;
  const bf16_t* p = Bs + nr * LDA + kb;
  union { v16bf v; v8bf h[2]; } f;
  f.h[0] = *(const v8bf*)(p);
  f.h[1] = *(const v8bf*)(p + 16);
  return f.v;
}

// Stage a 128x32 bf16 tile (global row stride F) into LDS [row][k], 256 thr
__device__ __forceinline__ void stage_k_tile(bf16_t* dst, const bf16_t* src, int tid) {
#pragma unroll
  for (int t = 0; t < 2; ++t) {
    const int j   = tid + t * 256;       // 0..511, 8 bf16 (16B) each
    const int row = j >> 2;
    const int ch  = (j & 3) * 8;
    cp_b128(dst + row * LDA + ch, src + (size_t)row * F_DIM + ch);
  }
}

// ---------------------------------------------------------------------------
// Kernel 0: zero the column-sum accumulator
// ---------------------------------------------------------------------------
__global__ void zero_kernel(float* __restrict__ c, int n) {
  const int i = blockIdx.x * 256 + threadIdx.x;
  if (i < n) c[i] = 0.f;
}

// ---------------------------------------------------------------------------
// Kernel 1: q/k/v projections.  GEMM [BT x F] * [F x F] with bf16 WMMA.
// grid = (BT/128, F/128, 3), block = 256 (8 waves).  q pre-scaled by 1/sqrt(F).
// ---------------------------------------------------------------------------
__global__ __launch_bounds__(256) void proj_kernel(
    const float* __restrict__ x,
    const float* __restrict__ Wq, const float* __restrict__ Wk,
    const float* __restrict__ Wv,
    bf16_t* __restrict__ qkv) {
  __shared__ bf16_t As[128 * LDA];
  __shared__ bf16_t Bs[128 * LDA];

  const int mblk = blockIdx.x;
  const int nblk = blockIdx.y;
  const int z    = blockIdx.z;
  const float* W = (z == 0) ? Wq : ((z == 1) ? Wk : Wv);
  bf16_t* outp   = qkv + (size_t)z * BTF;

  const int tid  = threadIdx.x;
  const int wave = tid >> 5;
  const int lane = tid & 31;
  const int wm   = wave & 3;     // 4 row groups of 32
  const int wn   = wave >> 2;    // 2 col groups of 64
  const size_t arow0 = (size_t)mblk * 128;
  const int    ncol0 = nblk * 128;

  v8f acc[2][4];
#pragma unroll
  for (int i = 0; i < 2; ++i)
#pragma unroll
    for (int j = 0; j < 4; ++j) acc[i][j] = zero8();

  for (int kt = 0; kt < F_DIM; kt += 32) {
    // --- stage A: x[arow0..+127][kt..+31] fp32 -> packed bf16 ---
#pragma unroll
    for (int t = 0; t < 4; ++t) {
      const int j   = tid + t * 256;          // 0..1023 float4s
      const int row = j >> 3;
      const int c4  = (j & 7) * 4;
      const float4 v = *(const float4*)(x + (arow0 + row) * F_DIM + kt + c4);
      unsigned* d = (unsigned*)(As + row * LDA + c4);
      d[0] = f2bf2(v.x, v.y);
      d[1] = f2bf2(v.z, v.w);
    }
    // --- stage B transposed: Bs[n][k] = W[kt+k][ncol0+n] fp32 -> bf16 ---
#pragma unroll
    for (int t = 0; t < 4; ++t) {
      const int j  = tid + t * 256;           // 0..1023 float4s
      const int kr = j >> 5;
      const int n4 = (j & 31) * 4;
      const float4 v = *(const float4*)(W + (size_t)(kt + kr) * F_DIM + ncol0 + n4);
      Bs[(n4 + 0) * LDA + kr] = f2bf(v.x);
      Bs[(n4 + 1) * LDA + kr] = f2bf(v.y);
      Bs[(n4 + 2) * LDA + kr] = f2bf(v.z);
      Bs[(n4 + 3) * LDA + kr] = f2bf(v.w);
    }
    __syncthreads();

    v16bf a0 = load_a_frag_s(As, LDA, wm * 32 + 0,  0, lane);
    v16bf a1 = load_a_frag_s(As, LDA, wm * 32 + 16, 0, lane);
#pragma unroll
    for (int j = 0; j < 4; ++j) {
      v16bf b = load_b_frag(Bs, wn * 64 + j * 16, lane);
      acc[0][j] = wmma_bf16(a0, b, acc[0][j]);
      acc[1][j] = wmma_bf16(a1, b, acc[1][j]);
    }
    __syncthreads();
  }

  // epilogue: fp32 acc -> bf16 store (q pre-scaled by 1/sqrt(F))
  const float scl    = (z == 0) ? SCALE_QK : 1.0f;
  const int   col    = ncol0 + wn * 64 + (lane & 15);
  const int   rowoff = (lane >> 4) * 8;
#pragma unroll
  for (int i = 0; i < 2; ++i)
#pragma unroll
    for (int j = 0; j < 4; ++j)
#pragma unroll
      for (int r = 0; r < 8; ++r) {
        const size_t row = arow0 + wm * 32 + i * 16 + rowoff + r;
        outp[row * F_DIM + col + j * 16] = f2bf(acc[i][j][r] * scl);
      }
}

// ---------------------------------------------------------------------------
// Kernel 2: softmax stats + attention column sums.
// grid = (T/128, B), block = 256 (8 waves), dynamic LDS = 284672 B:
//   [ Q resident: 128x1024 bf16, stride 1032 ][ K ping-pong: 2 x 128x32, stride 40 ]
// Q staged ONCE via async-to-LDS.  K tiles stream through a double buffer:
// the async copy of tile i+1 is issued before the 8 WMMAs of tile i, so the
// ASYNCcnt transfer overlaps the matrix math; one barrier per k-step.
// ---------------------------------------------------------------------------
__global__ __launch_bounds__(256) void attn_kernel(
    const bf16_t* __restrict__ qkv, float* __restrict__ csum) {
  extern __shared__ bf16_t smem[];
  bf16_t* Qs   = smem;                  // 128 * QLD
  bf16_t* Kbuf = smem + Q_ELEMS;        // 2 * KS_ELEMS

  const int tblk = blockIdx.x;
  const int b    = blockIdx.y;
  const bf16_t* q = qkv + (size_t)b * T_DIM * F_DIM;
  const bf16_t* k = qkv + BTF + (size_t)b * T_DIM * F_DIM;

  const int tid  = threadIdx.x;
  const int wave = tid >> 5;
  const int lane = tid & 31;
  const size_t qrow0 = (size_t)tblk * 128;

  // ---- stage the whole 128x1024 Q block into LDS once (async) ----
#pragma unroll 4
  for (int t = 0; t < 64; ++t) {
    const int j   = tid + t * 256;    // 16384 chunks of 8 bf16
    const int row = j >> 7;           // 128 chunks per row
    const int ch  = (j & 127) * 8;
    cp_b128(Qs + row * QLD + ch, q + (qrow0 + row) * F_DIM + ch);
  }
  wait_async();
  __syncthreads();

  float m_run[8], Zr[8];
#pragma unroll
  for (int r = 0; r < 8; ++r) { m_run[r] = -1e30f; Zr[r] = 0.f; }

  // -------- pass A: row max & normalizer --------
  for (int st = 0; st < T_DIM; st += 128) {
    const bf16_t* kbase = k + (size_t)st * F_DIM;
    if (st + 128 < T_DIM)
      __builtin_prefetch(k + (size_t)(st + 128) * F_DIM, 0, 2);
    v8f acc[8];
#pragma unroll
    for (int j = 0; j < 8; ++j) acc[j] = zero8();

    stage_k_tile(Kbuf, kbase, tid);                       // prime buffer 0
    for (int i = 0; i < F_DIM / 32; ++i) {
      const int kt = i * 32;
      bf16_t* cur = Kbuf + (i & 1) * KS_ELEMS;
      wait_async();
      __syncthreads();
      if (i + 1 < F_DIM / 32)                             // async copy overlaps WMMAs
        stage_k_tile(Kbuf + ((i + 1) & 1) * KS_ELEMS, kbase + kt + 32, tid);
      v16bf a = load_a_frag_s(Qs, QLD, wave * 16, kt, lane);
#pragma unroll
      for (int j = 0; j < 8; ++j) {
        v16bf bb = load_b_frag(cur, j * 16, lane);
        acc[j] = wmma_bf16(a, bb, acc[j]);
      }
    }

#pragma unroll
    for (int r = 0; r < 8; ++r) {
      float pm = acc[0][r];
#pragma unroll
      for (int j = 1; j < 8; ++j) pm = fmaxf(pm, acc[j][r]);
      pm = fmaxf(pm, __shfl_xor(pm, 1, 16));
      pm = fmaxf(pm, __shfl_xor(pm, 2, 16));
      pm = fmaxf(pm, __shfl_xor(pm, 4, 16));
      pm = fmaxf(pm, __shfl_xor(pm, 8, 16));
      const float mn = fmaxf(m_run[r], pm);
      float ls = 0.f;
#pragma unroll
      for (int j = 0; j < 8; ++j) ls += __expf(acc[j][r] - mn);
      ls += __shfl_xor(ls, 1, 16);
      ls += __shfl_xor(ls, 2, 16);
      ls += __shfl_xor(ls, 4, 16);
      ls += __shfl_xor(ls, 8, 16);
      Zr[r]    = Zr[r] * __expf(m_run[r] - mn) + ls;
      m_run[r] = mn;
    }
  }

  float rZ[8];
#pragma unroll
  for (int r = 0; r < 8; ++r) rZ[r] = 1.0f / Zr[r];

  // -------- pass B: recompute scores, accumulate column sums of attn --------
  for (int st = 0; st < T_DIM; st += 128) {
    const bf16_t* kbase = k + (size_t)st * F_DIM;
    if (st + 128 < T_DIM)
      __builtin_prefetch(k + (size_t)(st + 128) * F_DIM, 0, 2);
    v8f acc[8];
#pragma unroll
    for (int j = 0; j < 8; ++j) acc[j] = zero8();

    stage_k_tile(Kbuf, kbase, tid);
    for (int i = 0; i < F_DIM / 32; ++i) {
      const int kt = i * 32;
      bf16_t* cur = Kbuf + (i & 1) * KS_ELEMS;
      wait_async();
      __syncthreads();
      if (i + 1 < F_DIM / 32)
        stage_k_tile(Kbuf + ((i + 1) & 1) * KS_ELEMS, kbase + kt + 32, tid);
      v16bf a = load_a_frag_s(Qs, QLD, wave * 16, kt, lane);
#pragma unroll
      for (int j = 0; j < 8; ++j) {
        v16bf bb = load_b_frag(cur, j * 16, lane);
        acc[j] = wmma_bf16(a, bb, acc[j]);
      }
    }

#pragma unroll
    for (int j = 0; j < 8; ++j) {
      float cs = 0.f;
#pragma unroll
      for (int r = 0; r < 8; ++r)
        cs += __expf(acc[j][r] - m_run[r]) * rZ[r];
      cs += __shfl_xor(cs, 16, 32);            // combine row halves
      if (lane < 16)
        atomicAdd(&csum[b * T_DIM + st + j * 16 + lane], cs);
    }
  }
}

// ---------------------------------------------------------------------------
// Kernel 3: out[b,f] = (sum_s c[b,s]*v[b,s,f] + sum_t x[b,t,f]) / T
// ---------------------------------------------------------------------------
__global__ __launch_bounds__(256) void finalize_kernel(
    const float* __restrict__ x, const bf16_t* __restrict__ vb,
    const float* __restrict__ csum, float* __restrict__ out) {
  const int idx = blockIdx.x * 256 + threadIdx.x;   // 0 .. B*F-1
  const int b = idx >> 10;
  const int f = idx & (F_DIM - 1);
  float accv = 0.f, accx = 0.f;
  for (int s = 0; s < T_DIM; ++s) {
    const size_t row = (size_t)b * T_DIM + s;
    accv += csum[b * T_DIM + s] * bf2f(vb[row * F_DIM + f]);
    accx += x[row * F_DIM + f];
  }
  out[idx] = (accv + accx) * (1.0f / (float)T_DIM);
}

// ---------------------------------------------------------------------------
// Launch
// ---------------------------------------------------------------------------
extern "C" void kernel_launch(void* const* d_in, const int* in_sizes, int n_in,
                              void* d_out, int out_size, void* d_ws, size_t ws_size,
                              hipStream_t stream) {
  const float* x  = (const float*)d_in[0];
  const float* Wq = (const float*)d_in[1];
  const float* Wk = (const float*)d_in[2];
  const float* Wv = (const float*)d_in[3];
  float* out = (float*)d_out;

  // ws layout: [qkv bf16: 3*BTF*2 = 192MB][csum f32: B*T*4 = 128KB]
  bf16_t* qkv  = (bf16_t*)d_ws;
  float*  csum = (float*)((char*)d_ws + (size_t)3 * BTF * sizeof(bf16_t));

  (void)hipFuncSetAttribute((const void*)attn_kernel,
                            hipFuncAttributeMaxDynamicSharedMemorySize,
                            ATTN_SMEM_BYTES);

  zero_kernel<<<(BT_DIM + 255) / 256, 256, 0, stream>>>(csum, BT_DIM);

  dim3 pg(BT_DIM / 128, F_DIM / 128, 3);
  proj_kernel<<<pg, 256, 0, stream>>>(x, Wq, Wk, Wv, qkv);

  dim3 ag(T_DIM / 128, B_DIM);
  attn_kernel<<<ag, 256, ATTN_SMEM_BYTES, stream>>>(qkv, csum);

  finalize_kernel<<<(B_DIM * F_DIM) / 256, 256, 0, stream>>>(
      x, qkv + (size_t)2 * BTF, csum, out);
}
